// GCN_6244882448592
// MI455X (gfx1250) — matrix operand order repaired and checked by reading, not compiled, not run on hardware
//
#include <hip/hip_runtime.h>
#include <math.h>

// ---------------------------------------------------------------------------
// GAT pipeline for MI455X (gfx1250, wave32).  Heavy GEMMs use
// v_wmma_f32_16x16x32_bf16; edge softmax/aggregation uses L2 float atomics.
// ---------------------------------------------------------------------------

#define DIM 256      // hidden width (HEADS*HID for conv1, D for conv2)
#define NGRAPH 16

typedef __attribute__((ext_vector_type(16))) __bf16        v16bf;
typedef __attribute__((ext_vector_type(8)))  float         v8f;
typedef __attribute__((ext_vector_type(4)))  unsigned int  u32x4;

__device__ __forceinline__ unsigned short f2bf(float f) {
  unsigned int u = __float_as_uint(f);
  u += 0x7FFFu + ((u >> 16) & 1u);          // round-to-nearest-even
  return (unsigned short)(u >> 16);
}

__device__ __forceinline__ float gelu_f(float v) {
  float c = v + 0.044715f * v * v * v;
  return 0.5f * v * (1.0f + tanhf(0.7978845608028654f * c));
}

__device__ __forceinline__ void atomicMaxF(float* addr, float val) {
  if (val >= 0.0f) atomicMax((int*)addr, __float_as_int(val));
  else             atomicMin((unsigned int*)addr, __float_as_uint(val));
}

// ---------------------------------------------------------------------------
// Generic fill
// ---------------------------------------------------------------------------
__global__ __launch_bounds__(256) void fill_k(float* p, float v, int n) {
  int i = blockIdx.x * 256 + threadIdx.x;
  if (i < n) p[i] = v;
}

// ---------------------------------------------------------------------------
// Edge list with self loops; also emit attn_edge_index (as float) to output
// ---------------------------------------------------------------------------
__global__ __launch_bounds__(256) void build_edges_k(const int* __restrict__ ei,
                                                     int* srcA, int* dstA,
                                                     int E, int Et) {
  int e = blockIdx.x * 256 + threadIdx.x;
  if (e >= Et) return;
  if (e < E) { srcA[e] = ei[e]; dstA[e] = ei[E + e]; }
  else       { srcA[e] = e - E; dstA[e] = e - E; }
}

__global__ __launch_bounds__(256) void eiout_k(const int* __restrict__ srcA,
                                               const int* __restrict__ dstA,
                                               float* out, int Et) {
  int e = blockIdx.x * 256 + threadIdx.x;
  if (e >= Et) return;
  out[e]      = (float)srcA[e];
  out[Et + e] = (float)dstA[e];
}

// ---------------------------------------------------------------------------
// conv1: h = x @ W1 (outer product, in-features = 1), plus per-head
// attention scalars a_s/a_d (H=4, C=64).  One block per node.
// ---------------------------------------------------------------------------
__global__ __launch_bounds__(256) void conv1_node_k(const float* __restrict__ x,
                                                    const float* __restrict__ W1,
                                                    const float* __restrict__ as1,
                                                    const float* __restrict__ ad1,
                                                    float* __restrict__ hw,
                                                    float* __restrict__ a_s,
                                                    float* __restrict__ a_d) {
  __shared__ float sA[DIM];
  __shared__ float sB[DIM];
  int n = blockIdx.x;
  int t = threadIdx.x;
  float v = x[n] * W1[t];
  hw[(size_t)n * DIM + t] = v;
  sA[t] = v * as1[t];
  sB[t] = v * ad1[t];
  __syncthreads();
  if (t < 4) {
    float s = 0.f;
    for (int i = 0; i < 64; ++i) s += sA[t * 64 + i];
    a_s[n * 4 + t] = s;
  } else if (t < 8) {
    int h = t - 4;
    float s = 0.f;
    for (int i = 0; i < 64; ++i) s += sB[h * 64 + i];
    a_d[n * 4 + h] = s;
  }
}

// ---------------------------------------------------------------------------
// Weight transpose + bf16 convert:  Wt[n][k] = bf16(W[k][n])  (256x256)
// ---------------------------------------------------------------------------
__global__ __launch_bounds__(256) void wconv_k(const float* __restrict__ W,
                                               unsigned short* __restrict__ Wt) {
  int k = blockIdx.x;
  int n = threadIdx.x;
  Wt[(size_t)n * DIM + k] = f2bf(W[(size_t)k * DIM + n]);
}

// ---------------------------------------------------------------------------
// WMMA GEMM:  C[N,256] = A_bf16[N,256] @ B[256,256]  (Bt is B transposed,
// bf16).  Block = 128 threads (4 waves), each block covers a 16-row strip,
// each wave produces a 16x64 slab (4 accumulators), K-loop of 8 x k32.
// 32 v_wmma_f32_16x16x32_bf16 per wave.
// ---------------------------------------------------------------------------
__global__ __launch_bounds__(128) void gemm_bf16_wmma_k(
    const unsigned short* __restrict__ A,   // [N,256] bf16 row-major
    const unsigned short* __restrict__ Bt,  // [256,256] bf16, Bt[n][k]
    float* __restrict__ C) {
  const int lane = threadIdx.x & 31;
  const int wave = threadIdx.x >> 5;
  const int hi   = lane >> 4;               // 0 or 1 (half-wave)
  const int m0   = blockIdx.x * 16;
  const int n0   = wave * 64;
  const int mrow = m0 + (lane & 15);

  v8f acc[4];
#pragma unroll
  for (int t = 0; t < 4; ++t)
#pragma unroll
    for (int i = 0; i < 8; ++i) acc[t][i] = 0.0f;

#pragma unroll
  for (int kb = 0; kb < DIM; kb += 32) {
    // A fragment (ISA 16-bit A 16x32 layout):
    //   elems 0..7  -> K = kb + hi*8 + [0..7]
    //   elems 8..15 -> K = kb + hi*8 + 16 + [0..7]
    union { v16bf v; u32x4 u[2]; } af;
    const u32x4* ap = (const u32x4*)(A + (size_t)mrow * DIM + kb + hi * 8);
    af.u[0] = ap[0];
    af.u[1] = ap[2];   // +16 halfs = +2 * (8 halfs)

#pragma unroll
    for (int t = 0; t < 4; ++t) {
      // B fragment (32x16, 16-bit): lane n + 16*hi holds K = kb + hi*8 + [0..7], kb + hi*8 + 16 + [0..7]
      int ncol = n0 + t * 16 + (lane & 15);
      union { v16bf v; u32x4 u[2]; } bfv;
      const u32x4* bp = (const u32x4*)(Bt + (size_t)ncol * DIM + kb + hi * 8);
      bfv.u[0] = bp[0];
      bfv.u[1] = bp[2];
      acc[t] = __builtin_amdgcn_wmma_f32_16x16x32_bf16(
          false, af.v, false, bfv.v, (short)0, acc[t], false, false);
    }
  }

  // C/D layout: vgpr r -> row m0 + r + 8*hi, col = n-tile + (lane&15)
#pragma unroll
  for (int t = 0; t < 4; ++t) {
    int ncol = n0 + t * 16 + (lane & 15);
#pragma unroll
    for (int r = 0; r < 8; ++r) {
      int mr = m0 + r + hi * 8;
      C[(size_t)mr * DIM + ncol] = acc[t][r];
    }
  }
}

// ---------------------------------------------------------------------------
// Per-node attention scalars for conv2 (H=1): dot(h_row, att) via one wave
// per node, shuffle reduction (wave32).
// ---------------------------------------------------------------------------
__global__ __launch_bounds__(256) void node_att_k(const float* __restrict__ hw,
                                                  const float* __restrict__ as_v,
                                                  const float* __restrict__ ad_v,
                                                  float* __restrict__ a_s,
                                                  float* __restrict__ a_d, int N) {
  int wave = threadIdx.x >> 5;
  int lane = threadIdx.x & 31;
  int n = blockIdx.x * 8 + wave;
  if (n >= N) return;
  const float* row = hw + (size_t)n * DIM;
  float s = 0.f, d = 0.f;
#pragma unroll
  for (int j = 0; j < 8; ++j) {
    int c = lane + 32 * j;
    float v = row[c];
    s += v * as_v[c];
    d += v * ad_v[c];
  }
#pragma unroll
  for (int m = 16; m >= 1; m >>= 1) {
    s += __shfl_xor(s, m, 32);
    d += __shfl_xor(d, m, 32);
  }
  if (lane == 0) { a_s[n] = s; a_d[n] = d; }
}

// ---------------------------------------------------------------------------
// Edge softmax phase 1: e = leaky_relu(a_s[src]+a_d[dst]); segment max
// ---------------------------------------------------------------------------
__global__ __launch_bounds__(256) void edge_max_k(const int* __restrict__ srcA,
                                                  const int* __restrict__ dstA,
                                                  const float* __restrict__ a_s,
                                                  const float* __restrict__ a_d,
                                                  float* __restrict__ ebuf,
                                                  float* __restrict__ mbuf,
                                                  int Et, int H) {
  int i = blockIdx.x * 256 + threadIdx.x;
  if (i >= Et * H) return;
  int e = i / H, h = i - e * H;
  int s = srcA[e], d = dstA[e];
  float v = a_s[s * H + h] + a_d[d * H + h];
  v = v > 0.f ? v : 0.2f * v;
  ebuf[i] = v;
  atomicMaxF(&mbuf[d * H + h], v);
}

// Phase 2: ex = exp(e - m[dst]); segment sum -> denom
__global__ __launch_bounds__(256) void edge_expsum_k(const int* __restrict__ dstA,
                                                     float* __restrict__ ebuf,
                                                     const float* __restrict__ mbuf,
                                                     float* __restrict__ denom,
                                                     int Et, int H) {
  int i = blockIdx.x * 256 + threadIdx.x;
  if (i >= Et * H) return;
  int e = i / H, h = i - e * H;
  int d = dstA[e];
  float ex = __expf(ebuf[i] - mbuf[d * H + h]);
  ebuf[i] = ex;
  atomicAdd(&denom[d * H + h], ex);
}

// Phase 3: scatter-add msg = h[src]*alpha into outacc[dst].
// 32 threads/edge, 8 channels per thread.
__global__ __launch_bounds__(256) void scatter_k(const int* __restrict__ srcA,
                                                 const int* __restrict__ dstA,
                                                 const float* __restrict__ ebuf,
                                                 const float* __restrict__ denom,
                                                 const float* __restrict__ msg,
                                                 float* __restrict__ outacc,
                                                 int Et, int H, int C) {
  int tid = blockIdx.x * 256 + threadIdx.x;
  int e = tid >> 5;
  if (e >= Et) return;
  int cb = (tid & 31) * 8;
  int s = srcA[e], d = dstA[e];
  const float* srow = msg + (size_t)s * DIM;
  float* drow = outacc + (size_t)d * DIM;
#pragma unroll
  for (int j = 0; j < 8; ++j) {
    int c = cb + j;
    int h = c / C;
    float alpha = ebuf[(size_t)e * H + h] / (denom[(size_t)d * H + h] + 1e-16f);
    atomicAdd(&drow[c], srow[c] * alpha);
  }
}

// Last-layer alpha written to output (H=1)
__global__ __launch_bounds__(256) void alpha_out_k(const float* __restrict__ ebuf,
                                                   const float* __restrict__ denom,
                                                   const int* __restrict__ dstA,
                                                   float* __restrict__ out, int Et) {
  int e = blockIdx.x * 256 + threadIdx.x;
  if (e >= Et) return;
  out[e] = ebuf[e] / (denom[dstA[e]] + 1e-16f);
}

// bias + gelu; also produce bf16 copy for next WMMA GEMM
__global__ __launch_bounds__(256) void bias_gelu_k(const float* __restrict__ acc,
                                                   const float* __restrict__ bias,
                                                   float* __restrict__ h,
                                                   unsigned short* __restrict__ hbf,
                                                   int total) {
  int i = blockIdx.x * 256 + threadIdx.x;
  if (i >= total) return;
  float g = gelu_f(acc[i] + bias[i & (DIM - 1)]);
  h[i] = g;
  hbf[i] = f2bf(g);
}

// ---------------------------------------------------------------------------
// Global mean pool
// ---------------------------------------------------------------------------
__global__ __launch_bounds__(256) void pool_sum_k(const float* __restrict__ h,
                                                  const int* __restrict__ batch,
                                                  float* __restrict__ pool, int total) {
  int i = blockIdx.x * 256 + threadIdx.x;
  if (i >= total) return;
  int n = i >> 8, c = i & (DIM - 1);
  atomicAdd(&pool[(size_t)batch[n] * DIM + c], h[i]);
}

__global__ __launch_bounds__(256) void pool_cnt_k(const int* __restrict__ batch,
                                                  float* __restrict__ cnt, int N) {
  int n = blockIdx.x * 256 + threadIdx.x;
  if (n >= N) return;
  atomicAdd(&cnt[batch[n]], 1.0f);
}

// ---------------------------------------------------------------------------
// MLP head (256->128->64->1, gelu, sigmoid), single block.
// ---------------------------------------------------------------------------
__global__ __launch_bounds__(256) void head_k(const float* __restrict__ pool,
                                              const float* __restrict__ cnt,
                                              const float* __restrict__ Wl1,
                                              const float* __restrict__ bl1,
                                              const float* __restrict__ Wl2,
                                              const float* __restrict__ bl2,
                                              const float* __restrict__ Wl3,
                                              const float* __restrict__ bl3,
                                              float* __restrict__ out) {
  __shared__ float g[NGRAPH * 256];
  __shared__ float g1[NGRAPH * 128];
  __shared__ float g2[NGRAPH * 64];
  int t = threadIdx.x;
  for (int i = t; i < NGRAPH * 256; i += 256) g[i] = pool[i] / cnt[i >> 8];
  __syncthreads();
  for (int o = t; o < NGRAPH * 128; o += 256) {
    int r = o >> 7, c = o & 127;
    float s = bl1[c];
    for (int k = 0; k < 256; ++k) s += g[r * 256 + k] * Wl1[k * 128 + c];
    g1[o] = gelu_f(s);
  }
  __syncthreads();
  for (int o = t; o < NGRAPH * 64; o += 256) {
    int r = o >> 6, c = o & 63;
    float s = bl2[c];
    for (int k = 0; k < 128; ++k) s += g1[r * 128 + k] * Wl2[k * 64 + c];
    g2[o] = gelu_f(s);
  }
  __syncthreads();
  if (t < NGRAPH) {
    float s = bl3[0];
    for (int k = 0; k < 64; ++k) s += g2[t * 64 + k] * Wl3[k];
    out[t] = 1.0f / (1.0f + __expf(-s));
  }
}

// ---------------------------------------------------------------------------
// Host driver
// ---------------------------------------------------------------------------
extern "C" void kernel_launch(void* const* d_in, const int* in_sizes, int n_in,
                              void* d_out, int out_size, void* d_ws, size_t ws_size,
                              hipStream_t stream) {
  const float* x    = (const float*)d_in[0];
  const int*   ei   = (const int*)d_in[1];
  const int*   batch= (const int*)d_in[2];
  const float* W1   = (const float*)d_in[3];
  const float* as1  = (const float*)d_in[4];
  const float* ad1  = (const float*)d_in[5];
  const float* b1   = (const float*)d_in[6];
  const float* W2   = (const float*)d_in[7];
  const float* as2  = (const float*)d_in[8];
  const float* ad2  = (const float*)d_in[9];
  const float* b2   = (const float*)d_in[10];
  const float* Wl1  = (const float*)d_in[11];
  const float* bl1  = (const float*)d_in[12];
  const float* Wl2  = (const float*)d_in[13];
  const float* bl2  = (const float*)d_in[14];
  const float* Wl3  = (const float*)d_in[15];
  const float* bl3  = (const float*)d_in[16];

  const int N  = in_sizes[0];        // 20000
  const int E  = in_sizes[1] / 2;    // 320000
  const int Et = E + N;              // +self-loops

  // ---- workspace layout (256B aligned slices) ----
  char* p = (char*)d_ws;
  auto take = [&](size_t bytes) -> void* {
    void* r = (void*)p;
    p += (bytes + 255) & ~(size_t)255;
    return r;
  };
  int*            srcA  = (int*)take(sizeof(int) * (size_t)Et);
  int*            dstA  = (int*)take(sizeof(int) * (size_t)Et);
  float*          h     = (float*)take(sizeof(float) * (size_t)N * DIM);
  unsigned short* hbf   = (unsigned short*)take(sizeof(short) * (size_t)N * DIM);
  float*          hw    = (float*)take(sizeof(float) * (size_t)N * DIM);
  float*          outac = (float*)take(sizeof(float) * (size_t)N * DIM);
  float*          a_s   = (float*)take(sizeof(float) * (size_t)N * 4);
  float*          a_d   = (float*)take(sizeof(float) * (size_t)N * 4);
  float*          mbuf  = (float*)take(sizeof(float) * (size_t)N * 4);
  float*          denom = (float*)take(sizeof(float) * (size_t)N * 4);
  float*          ebuf  = (float*)take(sizeof(float) * (size_t)Et * 4);
  unsigned short* Wt    = (unsigned short*)take(sizeof(short) * DIM * DIM);
  float*          pool  = (float*)take(sizeof(float) * NGRAPH * DIM);
  float*          cnt   = (float*)take(sizeof(float) * NGRAPH);

  float* outF = (float*)d_out;
  float* out_ei    = outF + NGRAPH;            // [2, Et] as float
  float* out_alpha = outF + NGRAPH + 2 * Et;   // [Et]

  const float NEG_INF = -INFINITY;
  auto cdiv = [](int a, int b) { return (a + b - 1) / b; };

  // 1) edges (+self loops) and attn_edge_index output
  build_edges_k<<<cdiv(Et, 256), 256, 0, stream>>>(ei, srcA, dstA, E, Et);
  eiout_k<<<cdiv(Et, 256), 256, 0, stream>>>(srcA, dstA, out_ei, Et);

  // 2) conv1 (H=4, C=64)
  conv1_node_k<<<N, 256, 0, stream>>>(x, W1, as1, ad1, hw, a_s, a_d);
  fill_k<<<cdiv(N * 4, 256), 256, 0, stream>>>(mbuf, NEG_INF, N * 4);
  fill_k<<<cdiv(N * 4, 256), 256, 0, stream>>>(denom, 0.0f, N * 4);
  fill_k<<<cdiv(N * DIM, 256), 256, 0, stream>>>(outac, 0.0f, N * DIM);
  edge_max_k<<<cdiv(Et * 4, 256), 256, 0, stream>>>(srcA, dstA, a_s, a_d, ebuf, mbuf, Et, 4);
  edge_expsum_k<<<cdiv(Et * 4, 256), 256, 0, stream>>>(dstA, ebuf, mbuf, denom, Et, 4);
  scatter_k<<<cdiv(Et * 32, 256), 256, 0, stream>>>(srcA, dstA, ebuf, denom, hw, outac, Et, 4, 64);
  bias_gelu_k<<<cdiv(N * DIM, 256), 256, 0, stream>>>(outac, b1, h, hbf, N * DIM);

  // 3) three conv2 layers (H=1, C=256) — GEMM on WMMA
  for (int l = 0; l < 3; ++l) {
    wconv_k<<<DIM, DIM, 0, stream>>>(W2 + (size_t)l * DIM * DIM, Wt);
    gemm_bf16_wmma_k<<<N / 16, 128, 0, stream>>>(hbf, Wt, hw);
    node_att_k<<<cdiv(N, 8), 256, 0, stream>>>(hw, as2 + l * DIM, ad2 + l * DIM, a_s, a_d, N);
    fill_k<<<cdiv(N, 256), 256, 0, stream>>>(mbuf, NEG_INF, N);
    fill_k<<<cdiv(N, 256), 256, 0, stream>>>(denom, 0.0f, N);
    fill_k<<<cdiv(N * DIM, 256), 256, 0, stream>>>(outac, 0.0f, N * DIM);
    edge_max_k<<<cdiv(Et, 256), 256, 0, stream>>>(srcA, dstA, a_s, a_d, ebuf, mbuf, Et, 1);
    edge_expsum_k<<<cdiv(Et, 256), 256, 0, stream>>>(dstA, ebuf, mbuf, denom, Et, 1);
    if (l == 2)
      alpha_out_k<<<cdiv(Et, 256), 256, 0, stream>>>(ebuf, denom, dstA, out_alpha, Et);
    scatter_k<<<cdiv(Et * 32, 256), 256, 0, stream>>>(srcA, dstA, ebuf, denom, hw, outac, Et, 1, 256);
    bias_gelu_k<<<cdiv(N * DIM, 256), 256, 0, stream>>>(outac, b2 + l * DIM, h, hbf, N * DIM);
  }

  // 4) mean pool + MLP head
  fill_k<<<cdiv(NGRAPH * DIM, 256), 256, 0, stream>>>(pool, 0.0f, NGRAPH * DIM);
  fill_k<<<1, 256, 0, stream>>>(cnt, 0.0f, NGRAPH);
  pool_sum_k<<<cdiv(N * DIM, 256), 256, 0, stream>>>(h, batch, pool, N * DIM);
  pool_cnt_k<<<cdiv(N, 256), 256, 0, stream>>>(batch, cnt, N);
  head_k<<<1, 256, 0, stream>>>(pool, cnt, Wl1, bl1, Wl2, bl2, Wl3, bl3, outF);
}
